// Keypoint_9878424780874
// MI455X (gfx1250) — matrix operand order repaired and checked by reading, not compiled, not run on hardware
//
#include <hip/hip_runtime.h>
#include <hip/hip_bf16.h>

typedef __attribute__((ext_vector_type(16))) _Float16 v16h;
typedef __attribute__((ext_vector_type(8)))  float    v8f;

#define W_IMG 512
#define NPB   1024            // patches per batch (32*32)
#define BW_N  8
#define SCH   3
#define CCH   128
#define LOG2E 1.44269504088896340736f

// d_out layout (flat, in return order):
//   coords[BW*NPB*2] | scores[BW*SCH*NPB] | desc[BW*CCH*NPB]
#define COORDS_OFF 0
#define SCORES_OFF (BW_N*NPB*2)
#define DESC_OFF   (SCORES_OFF + BW_N*SCH*NPB)

// ---------------------------------------------------------------------------
// Kernel 1: per-patch softmax + expected coordinate via WMMA.
// One wave (32 threads) per block; each wave owns 16 consecutive patches.
// E(16x3) = exp_weights(16x256) x B(256x3), B = [u_local, v_local, 1].
// Patch staging uses GLOBAL_LOAD_ASYNC_TO_LDS_B128 (ASYNCcnt path).
// ---------------------------------------------------------------------------
__global__ __launch_bounds__(32)
void kp_softmax_expect_kernel(const float* __restrict__ det, float* __restrict__ out)
{
    const int lane = threadIdx.x;          // 0..31
    const int b    = blockIdx.x >> 6;      // batch
    const int tile = blockIdx.x & 63;      // 16-patch tile within batch

    __shared__ float s_patch[16 * 260];    // 16 patches x 256 (+4 pad), 16B-aligned stride
    __shared__ float s_max[16];
    __shared__ float s_res[48];            // [N][M], N=0..2, M=0..15

    const float* img = det + (size_t)b * W_IMG * W_IMG;

    // ---- async copy 16 patches into LDS (k = pi*16+pj unfold ordering) ----
    // 1024 16-byte segments total; 32 per lane. Both addresses 16B aligned.
    #pragma unroll 8
    for (int it = 0; it < 32; ++it) {
        int s  = it * 32 + lane;           // segment id 0..1023
        int m  = s >> 6;                   // patch within tile (64 segs/patch)
        int k  = (s & 63) << 2;            // first element index within patch
        int n  = tile * 16 + m;
        int pr = (n >> 5) << 4;            // patch base row
        int pc = (n & 31) << 4;            // patch base col
        int pi = k >> 4, pj = k & 15;      // pj multiple of 4 -> in-row 16B segment
        const float* gsrc = img + (size_t)(pr + pi) * W_IMG + pc + pj;
        unsigned lds_off  = (unsigned)(unsigned long long)(&s_patch[m * 260 + k]);
        asm volatile("global_load_async_to_lds_b128 %0, %1, off"
                     :: "v"(lds_off), "v"((unsigned long long)gsrc)
                     : "memory");
    }
    asm volatile("s_wait_asynccnt 0" ::: "memory");
    __syncthreads();

    // ---- per-patch max: 2 lanes per patch scan 128 values each ----
    {
        int m = lane >> 1;
        int h = lane & 1;
        float mv = -3.4e38f;
        #pragma unroll 4
        for (int i = 0; i < 128; ++i)
            mv = fmaxf(mv, s_patch[m * 260 + h * 128 + i]);
        mv = fmaxf(mv, __shfl_xor(mv, 1));
        if (h == 0) s_max[m] = mv;
    }
    __syncthreads();

    const int   mrow   = lane & 15;        // A row (patch) == B column id
    const int   lanehi = lane >> 4;        // half-wave select
    const float mx     = s_max[mrow];

    // ---- loop-invariant B pieces (branchless):
    //   within a 32-chunk: u_local = e (const), v_local = 2*kc + lanehi.
    //   B0[e] = e*m0 + lanehi*m1 + m2 ;  Vsel[e] = m1 ;  B = B0 + (2*kc)*Vsel
    const float m0 = (mrow == 0) ? 1.0f : 0.0f;
    const float m1 = (mrow == 1) ? 1.0f : 0.0f;
    const float m2 = (mrow == 2) ? 1.0f : 0.0f;
    v16h b0, vsel;
    #pragma unroll
    for (int e = 0; e < 16; ++e) {
        b0[e]   = (_Float16)((float)e * m0 + (float)lanehi * m1 + m2);
        vsel[e] = (_Float16)m1;
    }

    v8f acc = {};
    for (int kc = 0; kc < 8; ++kc) {
        // A: 16x32 f16 layout (contiguous k order, matches B):
        //   lane<16 : e=0..15 -> K=0..15
        //   lane>=16: e=0..15 -> K=16..31
        v16h a;
        #pragma unroll
        for (int e = 0; e < 16; ++e) {
            int kl = lanehi * 16 + e;
            int K  = kc * 32 + kl;
            float x = s_patch[mrow * 260 + K];
            a[e] = (_Float16)__builtin_amdgcn_exp2f((x - mx) * LOG2E);
        }
        v16h bm = b0 + (_Float16)(2 * kc) * vsel;   // packed f16 FMA, branchless
        acc = __builtin_amdgcn_wmma_f32_16x16x32_f16(
                /*neg_a=*/false, a, /*neg_b=*/false, bm,
                /*c_mod=*/(short)0, acc, /*reuse_a=*/false, /*reuse_b=*/false);
    }

    // ---- C/D layout: VGPR r, lanes 0-15 -> M=r (N=lane), lanes 16-31 -> M=r+8 ----
    if (lane <= 2) {
        #pragma unroll
        for (int r = 0; r < 8; ++r) s_res[lane * 16 + r] = acc[r];
    }
    if (lane >= 16 && lane <= 18) {
        #pragma unroll
        for (int r = 0; r < 8; ++r) s_res[(lane - 16) * 16 + 8 + r] = acc[r];
    }
    __syncthreads();

    if (lane < 16) {
        int m  = lane;
        int n  = tile * 16 + m;
        int pr = (n >> 5) << 4;
        int pc = (n & 31) << 4;
        float su  = s_res[m];
        float sv  = s_res[16 + m];
        float ss  = s_res[32 + m];
        float inv = 1.0f / ss;
        float u = su * inv + (float)pc;    // expected_u (x)
        float v = sv * inv + (float)pr;    // expected_v (y)
        size_t o = COORDS_OFF + ((size_t)(b * NPB + n)) * 2;
        out[o]     = u;
        out[o + 1] = v;
    }
}

// ---------------------------------------------------------------------------
// Kernel 2: bilinear gather of descriptors (128 ch) + weight scores (3 ch).
// grid_sample with align_corners=True and identical H=W for all tensors means
// the sample position is exactly the keypoint coordinate (u, v).
// One wave per channel handling 32 consecutive keypoints (coalesced writes).
// ---------------------------------------------------------------------------
__global__ __launch_bounds__(256)
void kp_sample_kernel(const float* __restrict__ wsc, const float* __restrict__ desc,
                      float* __restrict__ out)
{
    const int lane = threadIdx.x & 31;
    const int wave = threadIdx.x >> 5;
    const int c    = blockIdx.y * 8 + wave;     // 0..130 (128 desc + 3 score)
    const int n    = blockIdx.x * 32 + lane;    // keypoint index
    const int b    = blockIdx.z;
    if (c >= CCH + SCH) return;

    size_t co = COORDS_OFF + ((size_t)(b * NPB + n)) * 2;
    float x = out[co];
    float y = out[co + 1];

    float x0f = floorf(x), y0f = floorf(y);
    float wx1 = x - x0f,   wy1 = y - y0f;
    float wx0 = 1.0f - wx1, wy0 = 1.0f - wy1;
    int x0 = min(max((int)x0f,     0), W_IMG - 1);
    int x1 = min(max((int)x0f + 1, 0), W_IMG - 1);
    int y0 = min(max((int)y0f,     0), W_IMG - 1);
    int y1 = min(max((int)y0f + 1, 0), W_IMG - 1);

    const float* src;
    size_t oidx;
    if (c < CCH) {
        src  = desc + ((size_t)(b * CCH + c)) * W_IMG * W_IMG;
        oidx = DESC_OFF + ((size_t)(b * CCH + c)) * NPB + n;
    } else {
        int cs = c - CCH;
        src  = wsc + ((size_t)(b * SCH + cs)) * W_IMG * W_IMG;
        oidx = SCORES_OFF + ((size_t)(b * SCH + cs)) * NPB + n;
    }

    float v00 = src[(size_t)y0 * W_IMG + x0];
    float v01 = src[(size_t)y0 * W_IMG + x1];
    float v10 = src[(size_t)y1 * W_IMG + x0];
    float v11 = src[(size_t)y1 * W_IMG + x1];

    out[oidx] = v00 * (wy0 * wx0) + v01 * (wy0 * wx1)
              + v10 * (wy1 * wx0) + v11 * (wy1 * wx1);
}

// ---------------------------------------------------------------------------
extern "C" void kernel_launch(void* const* d_in, const int* in_sizes, int n_in,
                              void* d_out, int out_size, void* d_ws, size_t ws_size,
                              hipStream_t stream)
{
    (void)in_sizes; (void)n_in; (void)out_size; (void)d_ws; (void)ws_size;
    const float* det  = (const float*)d_in[0];   // (8,1,512,512)
    const float* wsc  = (const float*)d_in[1];   // (8,3,512,512)
    const float* desc = (const float*)d_in[2];   // (8,128,512,512)
    // d_in[3] = patch_size (int scalar) — fixed at 16, layouts hardcoded.
    float* out = (float*)d_out;

    // 8 batches * 64 tiles of 16 patches, one wave per tile
    kp_softmax_expect_kernel<<<dim3(BW_N * 64), dim3(32), 0, stream>>>(det, out);
    // 32 n-groups x 17 channel-groups (131 channels) x 8 batches
    kp_sample_kernel<<<dim3(32, 17, 8), dim3(256), 0, stream>>>(wsc, desc, out);
}
